// GPT2SymmetricLatentAttention_55181739819322
// MI455X (gfx1250) — matrix operand order repaired and checked by reading, not compiled, not run on hardware
//
#include <hip/hip_runtime.h>

#define DEVINL __device__ __forceinline__

typedef __attribute__((ext_vector_type(16))) _Float16 v16h;
typedef __attribute__((ext_vector_type(8)))  _Float16 v8h;
typedef __attribute__((ext_vector_type(8)))  float    v8f;

static constexpr int B_ = 4, T_ = 2048, C_ = 1024, H_ = 16, R_ = 64, HD_ = 64;

// --------------------------------------------------------------------------
// CDNA5 async global->LDS copy via inline asm (ISA 15.18.3, opcode 98).
// vdst = VGPR holding LDS byte address (addr[31:0] of generic shared ptr),
// vaddr = 64-bit global address pair, GV mode (saddr = off).
// --------------------------------------------------------------------------
DEVINL void async_load_lds_b128(const void* lds_dst, const void* gsrc) {
  unsigned lds_off = (unsigned)(unsigned long long)lds_dst;
  unsigned long long gaddr = (unsigned long long)gsrc;
  asm volatile("global_load_async_to_lds_b128 %0, %1, off"
               :
               : "v"(lds_off), "v"(gaddr)
               : "memory");
}
DEVINL void async_wait0() {
  asm volatile("s_wait_asynccnt 0" ::: "memory");
}

// ---------------------------------------------------------------------------
// WMMA fragment loaders (CDNA5 16-bit layouts, ISA 7.12.2)
// A 16x32: lane&15 = M row; lane>>4 selects K {0..7,16..23} vs {8..15,24..31}
// B 32x16: lane&15 = N row (of W[N,K]); lane>>4 selects K {0..7,16..23} vs {8..15,24..31}
// ---------------------------------------------------------------------------
DEVINL v16h wmma_load_a(const _Float16* p0, int ld, int lane) {
  const _Float16* p = p0 + (lane & 15) * ld + ((lane >> 4) << 3);
  v8h lo = *(const v8h*)(p);
  v8h hi = *(const v8h*)(p + 16);
  v16h r;
#pragma unroll
  for (int i = 0; i < 8; ++i) { r[i] = lo[i]; r[i + 8] = hi[i]; }
  return r;
}
DEVINL v16h wmma_load_b(const _Float16* p0, int ld, int lane) {
  const _Float16* p = p0 + (lane & 15) * ld + ((lane >> 4) << 3);
  v8h lo = *(const v8h*)(p);
  v8h hi = *(const v8h*)(p + 16);
  v16h r;
#pragma unroll
  for (int i = 0; i < 8; ++i) { r[i] = lo[i]; r[i + 8] = hi[i]; }
  return r;
}
DEVINL v8f wmma_f16(v16h a, v16h b, v8f c) {
  return __builtin_amdgcn_wmma_f32_16x16x32_f16(false, a, false, b, (short)0, c,
                                                false, false);
}

enum { SM_F16 = 0, SM_VT = 1, SM_Q = 2, SM_F32 = 3 };

// ---------------------------------------------------------------------------
// Small-N GEMM (N==64): 128 threads / 4 waves, 64x64 tile, direct loads.
// ---------------------------------------------------------------------------
template <int MODE, bool WSEL_Z>
__global__ __launch_bounds__(128) void gemm_small_kernel(
    const _Float16* __restrict__ A, const _Float16* __restrict__ W,
    _Float16* __restrict__ outH, int M, int N, int K) {
  const int lane = threadIdx.x & 31;
  const int wave = threadIdx.x >> 5;
  const int wm = blockIdx.x * 64 + (wave >> 1) * 32;
  const int wn = blockIdx.y * 64 + (wave & 1) * 32;
  const _Float16* Wb = W + (WSEL_Z ? (int)blockIdx.z * N * K : 0);

  v8f acc[2][2] = {{{}, {}}, {{}, {}}};
  for (int k = 0; k < K; k += 32) {
    if (k + 128 < K) {
      __builtin_prefetch(A + wm * K + k + 128, 0, 3);
      __builtin_prefetch(Wb + wn * K + k + 128, 0, 3);
    }
    v16h a0 = wmma_load_a(A + wm * K + k, K, lane);
    v16h a1 = wmma_load_a(A + (wm + 16) * K + k, K, lane);
    v16h b0 = wmma_load_b(Wb + wn * K + k, K, lane);
    v16h b1 = wmma_load_b(Wb + (wn + 16) * K + k, K, lane);
    acc[0][0] = wmma_f16(a0, b0, acc[0][0]);
    acc[0][1] = wmma_f16(a0, b1, acc[0][1]);
    acc[1][0] = wmma_f16(a1, b0, acc[1][0]);
    acc[1][1] = wmma_f16(a1, b1, acc[1][1]);
  }
  const int half = lane >> 4;
  const int nl = lane & 15;
#pragma unroll
  for (int i = 0; i < 2; ++i)
#pragma unroll
    for (int j = 0; j < 2; ++j)
#pragma unroll
      for (int e = 0; e < 8; ++e) {
        const int row = wm + 16 * i + e + 8 * half;
        const int col = wn + 16 * j + nl;
        const float v = acc[i][j][e];
        if (MODE == SM_F16) {
          outH[row * N + col] = (_Float16)v;
        } else {  // SM_Q: [B,H,T,R]
          const int b = row >> 11, t = row & (T_ - 1);
          outH[(((b * H_ + (int)blockIdx.z) * T_) + t) * R_ + col] = (_Float16)v;
        }
      }
}

// ---------------------------------------------------------------------------
// Big GEMM (N,K multiples of 128/32): 256 threads / 8 waves, 128x128 block
// tile staged through LDS with async global->LDS copies, wave tile 32x64
// => 8 WMMA per K=32 step per wave, staged data reused by 2-4 waves.
// ---------------------------------------------------------------------------
template <int MODE>
__global__ __launch_bounds__(256) void gemm_big_kernel(
    const _Float16* __restrict__ A, const _Float16* __restrict__ W,
    const float* __restrict__ bias, _Float16* __restrict__ outH,
    float* __restrict__ outF, int M, int N, int K) {
  __shared__ alignas(16) _Float16 As[128][40];
  __shared__ alignas(16) _Float16 Ws[128][40];
  const int tid = threadIdx.x;
  const int lane = tid & 31;
  const int wave = tid >> 5;
  const int bm = blockIdx.x * 128;
  const int bn = blockIdx.y * 128;
  const int wm_l = (wave >> 1) * 32;
  const int wn_l = (wave & 1) * 64;

  const int srow = tid >> 2;          // 0..63
  const int scol = (tid & 3) * 8;     // 0,8,16,24 (halves) = 16B chunks

  v8f acc[2][4];
#pragma unroll
  for (int i = 0; i < 2; ++i)
#pragma unroll
    for (int j = 0; j < 4; ++j) acc[i][j] = (v8f){};

  for (int k = 0; k < K; k += 32) {
    async_load_lds_b128(&As[srow][scol],      A + (bm + srow) * K + k + scol);
    async_load_lds_b128(&As[srow + 64][scol], A + (bm + srow + 64) * K + k + scol);
    async_load_lds_b128(&Ws[srow][scol],      W + (bn + srow) * K + k + scol);
    async_load_lds_b128(&Ws[srow + 64][scol], W + (bn + srow + 64) * K + k + scol);
    async_wait0();
    __syncthreads();
    v16h a0 = wmma_load_a(&As[wm_l][0], 40, lane);
    v16h a1 = wmma_load_a(&As[wm_l + 16][0], 40, lane);
    v16h b0 = wmma_load_b(&Ws[wn_l][0], 40, lane);
    v16h b1 = wmma_load_b(&Ws[wn_l + 16][0], 40, lane);
    v16h b2 = wmma_load_b(&Ws[wn_l + 32][0], 40, lane);
    v16h b3 = wmma_load_b(&Ws[wn_l + 48][0], 40, lane);
    acc[0][0] = wmma_f16(a0, b0, acc[0][0]);
    acc[0][1] = wmma_f16(a0, b1, acc[0][1]);
    acc[0][2] = wmma_f16(a0, b2, acc[0][2]);
    acc[0][3] = wmma_f16(a0, b3, acc[0][3]);
    acc[1][0] = wmma_f16(a1, b0, acc[1][0]);
    acc[1][1] = wmma_f16(a1, b1, acc[1][1]);
    acc[1][2] = wmma_f16(a1, b2, acc[1][2]);
    acc[1][3] = wmma_f16(a1, b3, acc[1][3]);
    __syncthreads();
  }

  const int half = lane >> 4;
  const int nl = lane & 15;
#pragma unroll
  for (int i = 0; i < 2; ++i)
#pragma unroll
    for (int j = 0; j < 4; ++j)
#pragma unroll
      for (int e = 0; e < 8; ++e) {
        const int row = bm + wm_l + 16 * i + e + 8 * half;
        const int col = bn + wn_l + 16 * j + nl;
        const float v = acc[i][j][e] + bias[col];
        if (MODE == SM_VT) {  // [B, C, T] transposed store for flash V
          const int b = row >> 11, t = row & (T_ - 1);
          outH[b * (N * T_) + col * T_ + t] = (_Float16)v;
        } else {  // SM_F32 final output
          outF[row * N + col] = v;
        }
      }
}

// ---------------------------------------------------------------------------
// Flash attention over the latent bilinear scores.
// Q = lt[b,h,:,R] (f16), K = latent[b,:,R] (f16), V = vT[b,h,HD,T] (f16).
// One block per (qtile64, h, b); each wave owns 16 query rows.
// ---------------------------------------------------------------------------
__global__ __launch_bounds__(128) void flash_latent_attn_kernel(
    const _Float16* __restrict__ lt, const _Float16* __restrict__ latent,
    const _Float16* __restrict__ vT, _Float16* __restrict__ y) {
  const int lane = threadIdx.x & 31;
  const int wave = threadIdx.x >> 5;
  const int h = blockIdx.y;
  const int b = blockIdx.z;
  const int qb = blockIdx.x * 64 + wave * 16;
  const int half = lane >> 4;
  const int nl = lane & 15;

  const _Float16* ltb = lt + ((b * H_ + h) * T_) * R_;
  const _Float16* latb = latent + (b * T_) * R_;
  const _Float16* vTb = vT + ((b * H_ + h) * HD_) * T_;

  const v16h q0 = wmma_load_a(ltb + qb * R_ + 0, R_, lane);
  const v16h q1 = wmma_load_a(ltb + qb * R_ + 32, R_, lane);

  v8f o0 = {}, o1 = {}, o2 = {}, o3 = {};
  float mrow[8], lrow[8];
#pragma unroll
  for (int e = 0; e < 8; ++e) { mrow[e] = -1e30f; lrow[e] = 0.f; }

  __shared__ alignas(16) _Float16 pl[4][16][40];  // per-wave P tile

  for (int kb = 0; kb <= qb + 15; kb += 32) {
    v8f s0 = {}, s1 = {};
    {
      v16h k00 = wmma_load_b(latb + kb * R_ + 0, R_, lane);
      v16h k01 = wmma_load_b(latb + kb * R_ + 32, R_, lane);
      s0 = wmma_f16(q0, k00, s0);
      s0 = wmma_f16(q1, k01, s0);
      v16h k10 = wmma_load_b(latb + (kb + 16) * R_ + 0, R_, lane);
      v16h k11 = wmma_load_b(latb + (kb + 16) * R_ + 32, R_, lane);
      s1 = wmma_f16(q0, k10, s1);
      s1 = wmma_f16(q1, k11, s1);
    }
    const int c0 = kb + nl, c1 = kb + 16 + nl;
#pragma unroll
    for (int e = 0; e < 8; ++e) {
      const int qrow = qb + e + 8 * half;
      float v0 = s0[e] * 0.125f;  // 1/sqrt(R)
      float v1 = s1[e] * 0.125f;
      if (c0 > qrow) v0 = -1e30f;  // causal mask
      if (c1 > qrow) v1 = -1e30f;
      float cm = fmaxf(v0, v1);
#pragma unroll
      for (int off = 1; off < 16; off <<= 1)
        cm = fmaxf(cm, __shfl_xor(cm, off, 32));
      const float nm = fmaxf(mrow[e], cm);
      const float alpha = __expf(mrow[e] - nm);
      mrow[e] = nm;
      const float p0 = __expf(v0 - nm);
      const float p1 = __expf(v1 - nm);
      float rs = p0 + p1;
#pragma unroll
      for (int off = 1; off < 16; off <<= 1) rs += __shfl_xor(rs, off, 32);
      lrow[e] = lrow[e] * alpha + rs;
      o0[e] *= alpha; o1[e] *= alpha; o2[e] *= alpha; o3[e] *= alpha;
      pl[wave][e + 8 * half][nl] = (_Float16)p0;
      pl[wave][e + 8 * half][16 + nl] = (_Float16)p1;
    }
    asm volatile("s_wait_dscnt 0" ::: "memory");
    const v16h pf = wmma_load_a(&pl[wave][0][0], 40, lane);
    const v16h vf0 = wmma_load_b(vTb + 0 * T_ + kb, T_, lane);
    const v16h vf1 = wmma_load_b(vTb + 16 * T_ + kb, T_, lane);
    const v16h vf2 = wmma_load_b(vTb + 32 * T_ + kb, T_, lane);
    const v16h vf3 = wmma_load_b(vTb + 48 * T_ + kb, T_, lane);
    o0 = wmma_f16(pf, vf0, o0);
    o1 = wmma_f16(pf, vf1, o1);
    o2 = wmma_f16(pf, vf2, o2);
    o3 = wmma_f16(pf, vf3, o3);
  }
#pragma unroll
  for (int e = 0; e < 8; ++e) {
    const float inv = 1.f / lrow[e];
    const int q = qb + e + 8 * half;
    const int rowoff = (b * T_ + q) * C_ + h * HD_;
    y[rowoff + 0 + nl]  = (_Float16)(o0[e] * inv);
    y[rowoff + 16 + nl] = (_Float16)(o1[e] * inv);
    y[rowoff + 32 + nl] = (_Float16)(o2[e] * inv);
    y[rowoff + 48 + nl] = (_Float16)(o3[e] * inv);
  }
}

// ---------------------------------------------------------------------------
// Small prep kernels
// ---------------------------------------------------------------------------
__global__ void cvt_f32_to_f16_kernel(const float* __restrict__ in,
                                      _Float16* __restrict__ out, int n) {
  const int i = blockIdx.x * 256 + threadIdx.x;
  if (i < n) out[i] = (_Float16)in[i];
}

__global__ void build_headmats_kernel(const float* __restrict__ core,
                                      const float* __restrict__ resid,
                                      _Float16* __restrict__ hmT) {
  const int i = blockIdx.x * 256 + threadIdx.x;  // < H*R*R = 65536
  const int h = i >> 12;
  const int rs = i & 4095;
  const int r = rs >> 6, s = rs & 63;
  float mean = 0.f;
#pragma unroll
  for (int hh = 0; hh < H_; ++hh) mean += resid[hh * 4096 + rs];
  mean *= (1.f / H_);
  const float val = 0.5f * (core[r * 64 + s] + core[s * 64 + r]) * (1.f / H_) +
                    (resid[h * 4096 + rs] - mean);
  hmT[(h << 12) + (s << 6) + r] = (_Float16)val;  // transposed for A@W^T GEMM
}

// ---------------------------------------------------------------------------
extern "C" void kernel_launch(void* const* d_in, const int* in_sizes, int n_in,
                              void* d_out, int out_size, void* d_ws,
                              size_t ws_size, hipStream_t stream) {
  (void)in_sizes; (void)n_in; (void)out_size; (void)ws_size;
  const float* hidden   = (const float*)d_in[0];
  const float* basis_w  = (const float*)d_in[1];
  const float* core     = (const float*)d_in[2];
  const float* head_res = (const float*)d_in[3];
  const float* v_w      = (const float*)d_in[4];
  const float* v_b      = (const float*)d_in[5];
  const float* o_w      = (const float*)d_in[6];
  const float* o_b      = (const float*)d_in[7];
  float* out = (float*)d_out;

  char* ws = (char*)d_ws;
  _Float16* hidden_h = (_Float16*)(ws + 0);         // [8192,1024]  16 MB
  _Float16* latent_h = (_Float16*)(ws + 16777216);  // [8192,64]     1 MB
  _Float16* lt_h     = (_Float16*)(ws + 17825792);  // [B,H,T,R]    16 MB
  _Float16* vT_h     = (_Float16*)(ws + 34603008);  // [B,H,HD,T]   16 MB
  _Float16* y_h      = (_Float16*)(ws + 51380224);  // [8192,1024]  16 MB
  _Float16* basis_h  = (_Float16*)(ws + 68157440);  // [64,1024]
  _Float16* vw_h     = (_Float16*)(ws + 68288512);  // [1024,1024]   2 MB
  _Float16* ow_h     = (_Float16*)(ws + 70385664);  // [1024,1024]   2 MB
  _Float16* hmT_h    = (_Float16*)(ws + 72482816);  // [H,R,R]

  cvt_f32_to_f16_kernel<<<8388608 / 256, 256, 0, stream>>>(hidden, hidden_h, 8388608);
  cvt_f32_to_f16_kernel<<<65536 / 256, 256, 0, stream>>>(basis_w, basis_h, 65536);
  cvt_f32_to_f16_kernel<<<1048576 / 256, 256, 0, stream>>>(v_w, vw_h, 1048576);
  cvt_f32_to_f16_kernel<<<1048576 / 256, 256, 0, stream>>>(o_w, ow_h, 1048576);
  build_headmats_kernel<<<65536 / 256, 256, 0, stream>>>(core, head_res, hmT_h);

  // latent = hidden @ basis_w^T            [8192,64]
  gemm_small_kernel<SM_F16, false><<<dim3(128, 1, 1), 128, 0, stream>>>(
      hidden_h, basis_h, latent_h, 8192, 64, 1024);
  // lt[b,h] = latent @ head_mats[h]        [B,H,T,R]
  gemm_small_kernel<SM_Q, true><<<dim3(128, 1, 16), 128, 0, stream>>>(
      latent_h, hmT_h, lt_h, 8192, 64, 64);
  // v = hidden @ v_w^T + v_b, stored transposed per head  [B,H,HD,T]
  gemm_big_kernel<SM_VT><<<dim3(64, 8, 1), 256, 0, stream>>>(
      hidden_h, vw_h, v_b, vT_h, nullptr, 8192, 1024, 1024);
  // y = softmax(causal(lt @ latent^T / sqrt(R))) @ v      [8192,1024]
  flash_latent_attn_kernel<<<dim3(32, 16, 4), 128, 0, stream>>>(lt_h, latent_h,
                                                                vT_h, y_h);
  // out = y @ o_w^T + o_b                                 [8192,1024] f32
  gemm_big_kernel<SM_F32><<<dim3(64, 8, 1), 256, 0, stream>>>(
      y_h, ow_h, o_b, nullptr, out, 8192, 1024, 1024);
}
